// MambaBlock_3212635537831
// MI455X (gfx1250) — compile-verified, matrix operand order checked by reading, not run on hardware
//
#include <hip/hip_runtime.h>

// ---------------------------------------------------------------------------
// Types for CDNA5 WMMA (wave32, 16x16x32 bf16 -> f32)
// ---------------------------------------------------------------------------
typedef __bf16 bf16_t;
typedef bf16_t bfx8  __attribute__((ext_vector_type(8)));
typedef bf16_t v16bf __attribute__((ext_vector_type(16)));
typedef float  v8f   __attribute__((ext_vector_type(8)));

union FragU { v16bf v; bfx8 h[2]; };

// ---------------------------------------------------------------------------
// gfx1250 async copy: global -> LDS, tracked by ASYNCcnt.
//   global_load_async_to_lds_b128 vDST(lds byte addr), v[ADDR] (64b global), off
// LDS address = low 32 bits of the generic pointer (aperture rules: generic
// LDS addresses are {SHARED_BASE, offset[31:0]}).
// ---------------------------------------------------------------------------
__device__ __forceinline__ void async_ld_b128(unsigned lds_addr, const void* gptr) {
    unsigned long long ga = (unsigned long long)(uintptr_t)gptr;
    asm volatile("global_load_async_to_lds_b128 %0, %1, off"
                 :: "v"(lds_addr), "v"(ga) : "memory");
}
__device__ __forceinline__ void wait_async_le6() {
    asm volatile("s_wait_asynccnt 0x6" ::: "memory");
}
__device__ __forceinline__ void wait_async_le0() {
    asm volatile("s_wait_asynccnt 0x0" ::: "memory");
}

// ---------------------------------------------------------------------------
// WMMA GEMM:  outF[M,N] = A[M,K](bf16) * W[N,K]^T(bf16)  (+ res)
//
// Block: 256 threads = 8 waves (2M x 4N). Block tile 128M x 256N.
// Wave tile 64M x 64N -> 4x4 v8f accumulators, 16 WMMAs / K-step of 32.
// Double-buffered LDS tiles filled with global_load_async_to_lds_b128
// (exactly 6 async instructions per wave per K-step, so
//  s_wait_asynccnt 6 retires precisely the previous slab; async loads
//  complete in order). LDS rows padded to 80B so fragment ds_load_b128s
// (rows r..r+15, stride 80B -> 20-bank step) cover all 64 banks conflict-free.
// ---------------------------------------------------------------------------
#define TILE_M 128
#define TILE_N 256
#define A_LDS_BYTES (128 * 80)                 // 10240
#define B_LDS_BYTES (256 * 80)                 // 20480
#define BUF_BYTES   (A_LDS_BYTES + B_LDS_BYTES) // 30720

__global__ __launch_bounds__(256)
void wmma_gemm_bf16(const bf16_t* __restrict__ A, const bf16_t* __restrict__ W,
                    float* __restrict__ outF, const float* __restrict__ res,
                    int M, int N, int K) {
    __shared__ __align__(16) char lds[2 * BUF_BYTES];   // 60 KB (WGP has 320 KB)
    const unsigned ldsBase0 = (unsigned)(uintptr_t)(&lds[0]);

    const int tid   = threadIdx.x;
    const int lane  = tid & 31;
    const int w     = tid >> 5;
    const int waveM = w & 1;
    const int waveN = w >> 1;
    const int mBlk  = blockIdx.y * TILE_M;
    const int nBlk  = blockIdx.x * TILE_N;
    const int m0    = mBlk + waveM * 64;
    const int n0    = nBlk + waveN * 64;
    const bool active = (n0 < N) && (m0 < M);
    const int nClampMax = (N - nBlk) - 1;      // clamp B-tile rows for padded N

    // Issue one K-slab (32 K) into LDS buffer `buf`: 6 async b128 per thread.
    auto issue_tile = [&](int buf, int kb) {
        unsigned base = ldsBase0 + (unsigned)buf * BUF_BYTES;
        #pragma unroll
        for (int i = 0; i < 2; ++i) {          // A: 128 rows x 4 chunks
            int c = tid + i * 256;
            int row = c >> 2, col = c & 3;
            async_ld_b128(base + row * 80u + col * 16u,
                          A + (size_t)(mBlk + row) * K + kb + col * 8);
        }
        #pragma unroll
        for (int i = 0; i < 4; ++i) {          // B: 256 rows x 4 chunks
            int c = tid + i * 256;
            int row = c >> 2, col = c & 3;
            int grow = row <= nClampMax ? row : nClampMax;
            async_ld_b128(base + A_LDS_BYTES + row * 80u + col * 16u,
                          W + (size_t)(nBlk + grow) * K + kb + col * 8);
        }
    };

    // CDNA5 16-bit A/B fragment layout (ISA 7.12.2): per lane two 16B chunks,
    //   lanes 0-15:  K+0..7  and K+16..23   (chunk 0, chunk 2)
    //   lanes 16-31: K+8..15 and K+24..31   (chunk 1, chunk 3)
    auto lds_frag = [&](unsigned tileBase, int rowBase) -> v16bf {
        unsigned r  = (unsigned)(rowBase + (lane & 15));
        unsigned c0 = (unsigned)(lane >> 4);
        const char* p = &lds[(tileBase - ldsBase0) + r * 80u + c0 * 16u];
        FragU f;
        f.h[0] = *(const bfx8*)(p);
        f.h[1] = *(const bfx8*)(p + 32);
        return f.v;
    };

    v8f acc[4][4];
    #pragma unroll
    for (int i = 0; i < 4; ++i)
        #pragma unroll
        for (int j = 0; j < 4; ++j)
            acc[i][j] = (v8f)(0.0f);

    const int nk = K >> 5;
    issue_tile(0, 0);
    for (int it = 0; it < nk; ++it) {
        const bool hasNext = (it + 1) < nk;
        if (hasNext) issue_tile((it + 1) & 1, (it + 1) << 5);
        if (hasNext) wait_async_le6(); else wait_async_le0();
        __syncthreads();                       // current slab visible to all waves

        const unsigned bufA = ldsBase0 + (unsigned)(it & 1) * BUF_BYTES;
        const unsigned bufB = bufA + A_LDS_BYTES;
        if (active) {                          // wave-uniform: execz branch skips WMMA
            v16bf af[4], bfr[4];
            #pragma unroll
            for (int mi = 0; mi < 4; ++mi) af[mi]  = lds_frag(bufA, waveM * 64 + mi * 16);
            #pragma unroll
            for (int ni = 0; ni < 4; ++ni) bfr[ni] = lds_frag(bufB, waveN * 64 + ni * 16);
            #pragma unroll
            for (int mi = 0; mi < 4; ++mi)
                #pragma unroll
                for (int ni = 0; ni < 4; ++ni)
                    acc[mi][ni] = __builtin_amdgcn_wmma_f32_16x16x32_bf16(
                        false, af[mi], false, bfr[ni], (short)0, acc[mi][ni], false, false);
        }
        __syncthreads();                       // all waves done reading this buffer
    }

    if (!active) return;
    // C/D layout: acc elem r -> (M = r + 8*(lane>>4), N = lane&15)
    const int mOff = (lane >> 4) << 3;
    const int nOff = lane & 15;
    if (res) {                                 // uniform: hoisted out of the loops
        #pragma unroll
        for (int mi = 0; mi < 4; ++mi) {
            #pragma unroll
            for (int ni = 0; ni < 4; ++ni) {
                int nCol = n0 + ni * 16 + nOff;
                #pragma unroll
                for (int r = 0; r < 8; ++r) {
                    int mRow = m0 + mi * 16 + mOff + r;
                    size_t idx = (size_t)mRow * (size_t)N + nCol;
                    outF[idx] = acc[mi][ni][r] + res[idx];
                }
            }
        }
    } else {
        #pragma unroll
        for (int mi = 0; mi < 4; ++mi) {
            #pragma unroll
            for (int ni = 0; ni < 4; ++ni) {
                int nCol = n0 + ni * 16 + nOff;
                #pragma unroll
                for (int r = 0; r < 8; ++r) {
                    int mRow = m0 + mi * 16 + mOff + r;
                    outF[(size_t)mRow * (size_t)N + nCol] = acc[mi][ni][r];
                }
            }
        }
    }
}

// ---------------------------------------------------------------------------
// RMSNorm over d_model, output bf16 for the WMMA in_proj GEMM.
// ---------------------------------------------------------------------------
__global__ __launch_bounds__(256)
void rmsnorm_kernel(const float* __restrict__ x, const float* __restrict__ nw,
                    bf16_t* __restrict__ out, int D) {
    int row = blockIdx.x;
    const float* xr = x + (size_t)row * D;
    float s = 0.f;
    for (int i = threadIdx.x; i < D; i += 256) { float v = xr[i]; s += v * v; }
    #pragma unroll
    for (int off = 16; off > 0; off >>= 1) s += __shfl_down(s, off, 32);
    __shared__ float red[8];
    __shared__ float rsh;
    if ((threadIdx.x & 31) == 0) red[threadIdx.x >> 5] = s;
    __syncthreads();
    if (threadIdx.x == 0) {
        float t = 0.f;
        #pragma unroll
        for (int i = 0; i < 8; ++i) t += red[i];
        rsh = rsqrtf(t / (float)D + 1e-5f);
    }
    __syncthreads();
    float r = rsh;
    for (int i = threadIdx.x; i < D; i += 256)
        out[(size_t)row * D + i] = (bf16_t)(xr[i] * r * nw[i]);
}

// ---------------------------------------------------------------------------
// f32 -> bf16 weight cast (grid-stride)
// ---------------------------------------------------------------------------
__global__ __launch_bounds__(256)
void cast_f32_bf16(const float* __restrict__ in, bf16_t* __restrict__ out, size_t n) {
    for (size_t i = blockIdx.x * 256ull + threadIdx.x; i < n; i += gridDim.x * 256ull)
        out[i] = (bf16_t)in[i];
}

// Pad x_proj_w (33 x 4096) -> (64 x 4096) bf16 so the WMMA tiler applies.
__global__ __launch_bounds__(256)
void pad_xproj_kernel(const float* __restrict__ w, bf16_t* __restrict__ out) {
    size_t i = blockIdx.x * 256ull + threadIdx.x;   // 64*4096 elements
    if (i >= 64ull * 4096ull) return;
    int row = (int)(i >> 12);
    out[i] = (row < 33) ? (bf16_t)w[i] : (bf16_t)0.f;
}

// ---------------------------------------------------------------------------
// Depthwise causal conv (K=4) + bias + SiLU. Writes f32 (scan input) and bf16
// (x_proj GEMM input). Coalesced over d.
// ---------------------------------------------------------------------------
__global__ __launch_bounds__(256)
void conv_silu_kernel(const float* __restrict__ xz, const float* __restrict__ cw,
                      const float* __restrict__ cb, float* __restrict__ xcF,
                      bf16_t* __restrict__ xcB) {
    int bl = blockIdx.x;           // b*2048 + l
    int l  = bl & 2047;
    for (int d = threadIdx.x; d < 4096; d += 256) {
        const float* wd = cw + (size_t)d * 4;
        float acc = cb[d];
        #pragma unroll
        for (int k = 0; k < 4; ++k) {
            int ls = l - 3 + k;
            if (ls >= 0) acc += wd[k] * xz[((size_t)(bl - 3 + k)) * 8192 + d];
        }
        float s = acc / (1.f + __expf(-acc));   // silu
        xcF[(size_t)bl * 4096 + d] = s;
        xcB[(size_t)bl * 4096 + d] = (bf16_t)s;
    }
}

// ---------------------------------------------------------------------------
// Selective scan: one lane per (b, d) channel, 16-state recurrence in regs,
// B/C/dt_raw broadcast per timestep through LDS. Fused with SiLU(z) gating;
// writes bf16 for the out_proj WMMA GEMM. 32 blocks x 8 waves = 256 waves.
// ---------------------------------------------------------------------------
__global__ __launch_bounds__(256)
void scan_gate_kernel(const float* __restrict__ xssm,   // (4096, 64): B|C|dt
                      const float* __restrict__ xcF,    // (4096, 4096) u
                      const float* __restrict__ xz,     // (4096, 8192) z at +4096
                      const float* __restrict__ dtw_p, const float* __restrict__ dtb_p,
                      const float* __restrict__ A_log, const float* __restrict__ Dp_p,
                      bf16_t* __restrict__ ygate) {
    int b = blockIdx.x >> 4;
    int d = ((blockIdx.x & 15) << 8) + threadIdx.x;
    float a[16], h[16];
    #pragma unroll
    for (int s = 0; s < 16; ++s) {
        a[s] = -__expf(A_log[(size_t)d * 16 + s]);
        h[s] = 0.f;
    }
    float dtw = dtw_p[d], dtb = dtb_p[d], Dp = Dp_p[d];
    __shared__ float bc[33];
    for (int t = 0; t < 2048; ++t) {
        size_t blt = (size_t)b * 2048 + t;
        __syncthreads();
        if (threadIdx.x < 33) bc[threadIdx.x] = xssm[blt * 64 + threadIdx.x];
        __syncthreads();
        float u   = xcF[blt * 4096 + d];
        float dtr = bc[32] * dtw + dtb;
        float dt  = (dtr > 20.f) ? dtr : log1pf(__expf(dtr));   // softplus
        float du  = dt * u;
        float y   = Dp * u;
        #pragma unroll
        for (int s = 0; s < 16; ++s) {
            float dA = __expf(dt * a[s]);
            h[s] = dA * h[s] + du * bc[s];
            y += h[s] * bc[16 + s];
        }
        float z = xz[blt * 8192 + 4096 + d];
        float g = y * (z / (1.f + __expf(-z)));                 // y * silu(z)
        ygate[blt * 4096 + d] = (bf16_t)g;
    }
}

// ---------------------------------------------------------------------------
// Launch
// ---------------------------------------------------------------------------
extern "C" void kernel_launch(void* const* d_in, const int* in_sizes, int n_in,
                              void* d_out, int out_size, void* d_ws, size_t ws_size,
                              hipStream_t stream) {
    (void)in_sizes; (void)n_in; (void)out_size; (void)ws_size;
    const float* x          = (const float*)d_in[0];
    const float* norm_w     = (const float*)d_in[1];
    const float* in_proj_w  = (const float*)d_in[2];
    const float* conv_w     = (const float*)d_in[3];
    const float* conv_b     = (const float*)d_in[4];
    const float* x_proj_w   = (const float*)d_in[5];
    const float* dt_w       = (const float*)d_in[6];
    const float* dt_b       = (const float*)d_in[7];
    const float* A_log      = (const float*)d_in[8];
    const float* D_param    = (const float*)d_in[9];
    const float* out_proj_w = (const float*)d_in[10];
    float* out = (float*)d_out;

    const int BL = 4096;       // B*L
    const int DM = 2048;       // d_model
    const int DI = 4096;       // d_inner
    const int DI2 = 8192;

    char* ws = (char*)d_ws;
    size_t off = 0;
    auto alloc = [&](size_t bytes) -> char* {
        char* p = ws + off;
        off = (off + bytes + 255) & ~(size_t)255;
        return p;
    };
    bf16_t* xn   = (bf16_t*)alloc((size_t)BL * DM * 2);        // 16 MB
    bf16_t* wip  = (bf16_t*)alloc((size_t)DI2 * DM * 2);       // 32 MB
    float*  xz   = (float*) alloc((size_t)BL * DI2 * 4);       // 128 MB
    bf16_t* wxp  = (bf16_t*)alloc((size_t)64 * DI * 2);        // 0.5 MB
    float*  xcF  = (float*) alloc((size_t)BL * DI * 4);        // 64 MB
    bf16_t* xcB  = (bf16_t*)alloc((size_t)BL * DI * 2);        // 32 MB
    float*  xssm = (float*) alloc((size_t)BL * 64 * 4);        // 1 MB
    bf16_t* yg   = (bf16_t*)alloc((size_t)BL * DI * 2);        // 32 MB
    bf16_t* wop  = (bf16_t*)alloc((size_t)DM * DI * 2);        // 16 MB

    // 1) RMSNorm -> bf16
    rmsnorm_kernel<<<BL, 256, 0, stream>>>(x, norm_w, xn, DM);
    // 2) weight casts
    cast_f32_bf16<<<8192, 256, 0, stream>>>(in_proj_w, wip, (size_t)DI2 * DM);
    cast_f32_bf16<<<8192, 256, 0, stream>>>(out_proj_w, wop, (size_t)DM * DI);
    pad_xproj_kernel<<<1024, 256, 0, stream>>>(x_proj_w, wxp);
    // 3) in_proj GEMM: (4096x2048) * (8192x2048)^T -> xz f32
    {
        dim3 grid(DI2 / TILE_N, BL / TILE_M);
        wmma_gemm_bf16<<<grid, 256, 0, stream>>>(xn, wip, xz, nullptr, BL, DI2, DM);
    }
    // 4) depthwise causal conv + SiLU
    conv_silu_kernel<<<BL, 256, 0, stream>>>(xz, conv_w, conv_b, xcF, xcB);
    // 5) x_proj GEMM (N padded 33 -> 64): (4096x4096) * (64x4096)^T -> xssm
    {
        dim3 grid(1, BL / TILE_M);
        wmma_gemm_bf16<<<grid, 256, 0, stream>>>(xcB, wxp, xssm, nullptr, BL, 64, DI);
    }
    // 6) selective scan + SiLU(z) gating -> yg bf16
    scan_gate_kernel<<<32, 256, 0, stream>>>(xssm, xcF, xz, dt_w, dt_b, A_log,
                                             D_param, yg);
    // 7) out_proj GEMM + residual: (4096x4096) * (2048x4096)^T + x -> out
    {
        dim3 grid(DM / TILE_N, BL / TILE_M);
        wmma_gemm_bf16<<<grid, 256, 0, stream>>>(yg, wop, out, x, BL, DM, DI);
    }
}
